// Starcoder2Attention_40218073759772
// MI455X (gfx1250) — matrix operand order repaired and checked by reading, not compiled, not run on hardware
//
#include <hip/hip_runtime.h>
#include <math.h>

// ---------------------------------------------------------------------------
// Starcoder2 attention layer for gfx1250 (MI455X), bf16 WMMA pipeline.
// B=2, S=2048, H=4608, NH=36, NKV=4, D=128
// ---------------------------------------------------------------------------

typedef __attribute__((ext_vector_type(16))) __bf16 v16bf;
typedef __attribute__((ext_vector_type(8)))  float  v8f;

#define NB 2
#define NS 2048
#define NHID 4608
#define NHEADS 36
#define NKVH 4
#define HDIM 128
#define GRP (NHEADS / NKVH)           // 9
#define QSZ (NHEADS * HDIM)           // 4608
#define KVSZ (NKVH * HDIM)            // 512
#define OQKV (QSZ + 2 * KVSZ)         // 5632
#define ATTN_SCALE 0.08838834764831845f   // 128^-0.5

// ---------------- CDNA5 async global->LDS path (guarded) -------------------
#if defined(__gfx1250__) && \
    __has_builtin(__builtin_amdgcn_global_load_async_to_lds_b128) && \
    __has_builtin(__builtin_amdgcn_s_wait_asynccnt)
#define HAVE_ASYNC_LDS 1
#else
#define HAVE_ASYNC_LDS 0
#endif

#define ASLDS __attribute__((address_space(3)))
#define ASGLB __attribute__((address_space(1)))

typedef __attribute__((__vector_size__(4 * sizeof(int)))) int i4vec;

#if HAVE_ASYNC_LDS
// 16B per lane, VGPR-free DMA into LDS, tracked by ASYNCcnt.
// Builtin signature (from clang diagnostic): (int4 AS1*, int4 AS3*, Ii, Ii)
#define ASYNC_CP16(gsrc, ldst)                                                 \
    __builtin_amdgcn_global_load_async_to_lds_b128(                            \
        (ASGLB i4vec*)(ASGLB void*)(void*)(const void*)(gsrc),                 \
        (ASLDS i4vec*)(ASLDS void*)(void*)(ldst), 0, 0)
#define ASYNC_WAIT() __builtin_amdgcn_s_wait_asynccnt(0)
#else
#define ASYNC_CP16(gsrc, ldst) (*(uint4*)(ldst) = *(const uint4*)(gsrc))
#define ASYNC_WAIT()
#endif

// ---------------- bf16 helpers (round-to-nearest-even) ---------------------
__device__ __forceinline__ unsigned short f2bf(float f) {
    union { float f; unsigned u; } v; v.f = f;
    unsigned r = v.u + 0x7FFFu + ((v.u >> 16) & 1u);
    return (unsigned short)(r >> 16);
}

union FragBF { v16bf v; unsigned u[8]; };

// A-matrix fragment (16x32 bf16). ISA 7.12.2: lanes 0-15 M=0..15 K=0..7,16..23;
// lanes 16-31 M=0..15 K=8..15,24..31.  K(p) = (p<4?0:16) + half*8 + 2*(p&3)
__device__ __forceinline__ v16bf load_frag_a(const unsigned short* base, int row,
                                             int ldk, int kof, int half) {
    FragBF f;
    const unsigned* b32 = (const unsigned*)(base + row * ldk + kof);
#pragma unroll
    for (int p = 0; p < 8; ++p) {
        int K = ((p < 4) ? 0 : 16) + half * 8 + 2 * (p & 3);
        f.u[p] = b32[K >> 1];
    }
    return f.v;
}

// B-matrix fragment (32x16 bf16), element (K,N) = Bsrc[row=N][K].
// Same K order as A: lanes 0-15 K=0..7,16..23; lanes 16-31 K=8..15,24..31,
// VGPR p holds K = (p<4?0:16) + half*8 + 2*(p&3), +1.
__device__ __forceinline__ v16bf load_frag_b(const unsigned short* base, int row,
                                             int ldk, int kof, int half) {
    FragBF f;
    const unsigned* b32 = (const unsigned*)(base + row * ldk + kof);
#pragma unroll
    for (int p = 0; p < 8; ++p) {
        int K = ((p < 4) ? 0 : 16) + half * 8 + 2 * (p & 3);
        f.u[p] = b32[K >> 1];
    }
    return f.v;
}

__device__ __forceinline__ v8f wmma_bf16(v16bf a, v16bf b, v8f c) {
    return __builtin_amdgcn_wmma_f32_16x16x32_bf16(false, a, false, b,
                                                   (short)0, c, false, false);
}

// ---------------------- f32 -> bf16 convert --------------------------------
__global__ void convert_f32_bf16(const float* __restrict__ src,
                                 unsigned short* __restrict__ dst, size_t n) {
    size_t i = (size_t)blockIdx.x * blockDim.x + threadIdx.x;
    size_t stride = (size_t)gridDim.x * blockDim.x;
    for (; i < n; i += stride) dst[i] = f2bf(src[i]);
}

// ---------------------- generic bf16 GEMM ----------------------------------
// C[M x N] = A[M x K](bf16) * W[N x K]^T(bf16) + bias[N], C f32.
// Block: 256 thr (8 waves). Block tile 128x64, K-step 32.
// Wave w -> rows [w*16, w*16+16), all 64 cols (4 accumulators).
#define GBM 128
#define GBN 64
#define GBK 32
__global__ __launch_bounds__(256) void gemm_bf16_kernel(
    const unsigned short* __restrict__ A, const unsigned short* __restrict__ W,
    const float* __restrict__ bias, float* __restrict__ C,
    int M, int N, int K) {
    __shared__ unsigned short As[GBM * GBK];   // 8 KB
    __shared__ unsigned short Bs[GBN * GBK];   // 4 KB
    int tid = threadIdx.x;
    int wave = tid >> 5, lane = tid & 31;
    int half = lane >> 4, lc = lane & 15;
    int rowBase = blockIdx.y * GBM;
    int colBase = blockIdx.x * GBN;

    v8f acc[4];
#pragma unroll
    for (int t = 0; t < 4; ++t)
#pragma unroll
        for (int r = 0; r < 8; ++r) acc[t][r] = 0.0f;

    for (int k0 = 0; k0 < K; k0 += GBK) {
        {   // A tile: 128x32, 2x16B async per thread
            int r = tid >> 1, c16 = (tid & 1) * 16;
            const unsigned short* g = A + (size_t)(rowBase + r) * K + k0 + c16;
            unsigned short* l = As + r * GBK + c16;
            ASYNC_CP16(g, l);
            ASYNC_CP16(g + 8, l + 8);
        }
        {   // B tile: 64x32, 1x16B async per thread
            int n = tid >> 2, c8 = (tid & 3) * 8;
            ASYNC_CP16(W + (size_t)(colBase + n) * K + k0 + c8,
                       Bs + n * GBK + c8);
        }
        if (k0 + GBK < K) {   // prefetch next A tile line -> global_prefetch
            int r = tid >> 1;
            __builtin_prefetch(A + (size_t)(rowBase + r) * K + k0 + GBK, 0, 3);
        }
        ASYNC_WAIT();
        __syncthreads();

        v16bf a = load_frag_a(As, wave * 16 + lc, GBK, 0, half);
#pragma unroll
        for (int t = 0; t < 4; ++t) {
            v16bf b = load_frag_b(Bs, t * 16 + lc, GBK, 0, half);
            acc[t] = wmma_bf16(a, b, acc[t]);
        }
        __syncthreads();
    }

#pragma unroll
    for (int t = 0; t < 4; ++t) {
        int col = colBase + t * 16 + lc;
        float bv = bias ? bias[col] : 0.0f;
#pragma unroll
        for (int r = 0; r < 8; ++r) {
            int row = rowBase + wave * 16 + r + 8 * half;
            C[(size_t)row * N + col] = acc[t][r] + bv;
        }
    }
}

// ---------------------- RoPE + QKV split -----------------------------------
// qkv f32 (B*S, 5632) -> q bf16 (B,S,NH,D) rope'd, k bf16 (B,S,NKV,D) rope'd,
// v bf16 (B,S,NKV,D).
__global__ void rope_split_kernel(const float* __restrict__ qkv,
                                  const int* __restrict__ positions,
                                  unsigned short* __restrict__ qout,
                                  unsigned short* __restrict__ kout,
                                  unsigned short* __restrict__ vout) {
    const int QP = NHEADS * (HDIM / 2);       // 2304 q rope-pairs / row
    const int KP = NKVH * (HDIM / 2);         // 256  k rope-pairs / row
    const int VE = NKVH * HDIM;               // 512  v elems / row
    const int perRow = QP + KP + VE;          // 3072
    size_t n = (size_t)NB * NS * perRow;
    size_t i = (size_t)blockIdx.x * blockDim.x + threadIdx.x;
    size_t stride = (size_t)gridDim.x * blockDim.x;
    const float LOG_THETA = 11.512925464970229f;  // ln(1e5)
    for (; i < n; i += stride) {
        size_t row = i / perRow;
        int o = (int)(i - row * perRow);
        int s = (int)(row % NS);
        const float* rowp = qkv + row * OQKV;
        float pos = (float)positions[s];
        if (o < QP) {
            int h = o / (HDIM / 2), j = o % (HDIM / 2);
            float inv = __expf(-(2.0f * j / HDIM) * LOG_THETA);
            float ang = pos * inv, c = __cosf(ang), sn = __sinf(ang);
            float x1 = rowp[h * HDIM + j];
            float x2 = rowp[h * HDIM + j + HDIM / 2];
            size_t base = (row * NHEADS + h) * HDIM;
            qout[base + j] = f2bf(x1 * c - x2 * sn);
            qout[base + j + HDIM / 2] = f2bf(x2 * c + x1 * sn);
        } else if (o < QP + KP) {
            int oo = o - QP;
            int h = oo / (HDIM / 2), j = oo % (HDIM / 2);
            float inv = __expf(-(2.0f * j / HDIM) * LOG_THETA);
            float ang = pos * inv, c = __cosf(ang), sn = __sinf(ang);
            float x1 = rowp[QSZ + h * HDIM + j];
            float x2 = rowp[QSZ + h * HDIM + j + HDIM / 2];
            size_t base = (row * NKVH + h) * HDIM;
            kout[base + j] = f2bf(x1 * c - x2 * sn);
            kout[base + j + HDIM / 2] = f2bf(x2 * c + x1 * sn);
        } else {
            int oo = o - QP - KP;
            vout[row * KVSZ + oo] = f2bf(rowp[QSZ + KVSZ + oo]);
        }
    }
}

// ---------------------- flash attention ------------------------------------
// Grid: (S/128, NH, B). Block 256 thr (8 waves). Each wave owns 16 q rows,
// online softmax over 32-key tiles. Out f32 (B,S,NH*D).
__global__ __launch_bounds__(256) void attn_kernel(
    const unsigned short* __restrict__ Qb, const unsigned short* __restrict__ Kb,
    const unsigned short* __restrict__ Vb, float* __restrict__ Out) {
    __shared__ unsigned short Qs[128 * 128];        // 32 KB  [qrow][d]
    __shared__ unsigned short Ks[32 * 128];         // 8 KB   [key][d]
    __shared__ unsigned short Vt[128 * 32];         // 8 KB   [d][key]
    __shared__ unsigned short Ps[8][16 * 32];       // 8 KB   per-wave P stage

    int tid = threadIdx.x, wave = tid >> 5, lane = tid & 31;
    int half = lane >> 4, lc = lane & 15;
    int qbase = blockIdx.x * 128;
    int h = blockIdx.y;
    int b = blockIdx.z;
    int kvh = h / GRP;

    {   // Q tile: 128 rows x 128 d, 8x16B async per thread
        int r = tid >> 1, d0 = (tid & 1) * 64;
        size_t g = ((size_t)(b * NS + qbase + r) * NHEADS + h) * HDIM + d0;
        const unsigned short* src = Qb + g;
        unsigned short* dst = Qs + r * 128 + d0;
#pragma unroll
        for (int i = 0; i < 8; ++i) ASYNC_CP16(src + i * 8, dst + i * 8);
    }

    v8f O[8];
    float mrow[8], lrow[8];
#pragma unroll
    for (int t = 0; t < 8; ++t)
#pragma unroll
        for (int r = 0; r < 8; ++r) O[t][r] = 0.0f;
#pragma unroll
    for (int r = 0; r < 8; ++r) { mrow[r] = -INFINITY; lrow[r] = 0.0f; }

    int nkt = (qbase + 128) / 32;
    for (int kt = 0; kt < nkt; ++kt) {
        int kbase = kt * 32;
        {   // K tile 32x128, async
            int r = tid >> 3, d0 = (tid & 7) * 16;
            size_t g = ((size_t)(b * NS + kbase + r) * NKVH + kvh) * HDIM + d0;
            const unsigned short* src = Kb + g;
            unsigned short* dst = Ks + r * 128 + d0;
            ASYNC_CP16(src, dst);
            ASYNC_CP16(src + 8, dst + 8);
        }
        {   // V tile 32x128, transposed into Vt[d][key] (16-bit scatter)
            int r = tid >> 3, d0 = (tid & 7) * 16;
            size_t g = ((size_t)(b * NS + kbase + r) * NKVH + kvh) * HDIM + d0;
            const unsigned short* src = Vb + g;
#pragma unroll
            for (int i = 0; i < 16; ++i) Vt[(d0 + i) * 32 + r] = src[i];
        }
        ASYNC_WAIT();
        __syncthreads();

        if (kbase <= qbase + wave * 16 + 15) {   // wave-uniform causal skip
            // ---- scores: Q(16x128) x K^T(128x32) -> 16x32
            v8f c0, c1;
#pragma unroll
            for (int r = 0; r < 8; ++r) { c0[r] = 0.0f; c1[r] = 0.0f; }
#pragma unroll
            for (int dk = 0; dk < 4; ++dk) {
                v16bf a  = load_frag_a(Qs, wave * 16 + lc, 128, dk * 32, half);
                v16bf b0 = load_frag_b(Ks, lc,      128, dk * 32, half);
                v16bf b1 = load_frag_b(Ks, 16 + lc, 128, dk * 32, half);
                c0 = wmma_bf16(a, b0, c0);
                c1 = wmma_bf16(a, b1, c1);
            }
            // ---- online softmax (per row; rows split across wave halves)
            float p0[8], p1[8];
#pragma unroll
            for (int r = 0; r < 8; ++r) {
                int qrow = qbase + wave * 16 + r + 8 * half;
                int key0 = kbase + lc, key1 = kbase + 16 + lc;
                float s0 = (key0 <= qrow) ? c0[r] * ATTN_SCALE : -INFINITY;
                float s1 = (key1 <= qrow) ? c1[r] * ATTN_SCALE : -INFINITY;
                float tm = fmaxf(s0, s1);
#pragma unroll
                for (int m = 1; m < 16; m <<= 1)
                    tm = fmaxf(tm, __shfl_xor(tm, m, 16));
                float nm = fmaxf(mrow[r], tm);
                float al = __expf(mrow[r] - nm);
                p0[r] = __expf(s0 - nm);
                p1[r] = __expf(s1 - nm);
                float ts = p0[r] + p1[r];
#pragma unroll
                for (int m = 1; m < 16; m <<= 1)
                    ts += __shfl_xor(ts, m, 16);
                lrow[r] = lrow[r] * al + ts;
                mrow[r] = nm;
#pragma unroll
                for (int t = 0; t < 8; ++t) O[t][r] *= al;
            }
            // ---- stage P (C-layout regs -> A-layout via per-wave LDS)
#pragma unroll
            for (int r = 0; r < 8; ++r) {
                int m = r + 8 * half;
                Ps[wave][m * 32 + lc]      = f2bf(p0[r]);
                Ps[wave][m * 32 + 16 + lc] = f2bf(p1[r]);
            }
            v16bf pa = load_frag_a(&Ps[wave][0], lc, 32, 0, half);
            // ---- O += P(16x32) x V(32x128)
#pragma unroll
            for (int t = 0; t < 8; ++t) {
                v16bf vb = load_frag_b(Vt, t * 16 + lc, 32, 0, half);
                O[t] = wmma_bf16(pa, vb, O[t]);
            }
        }
        __syncthreads();
    }

#pragma unroll
    for (int t = 0; t < 8; ++t)
#pragma unroll
        for (int r = 0; r < 8; ++r) {
            int qrow = qbase + wave * 16 + r + 8 * half;
            int d = t * 16 + lc;
            Out[((size_t)(b * NS + qrow) * NHEADS + h) * HDIM + d] =
                O[t][r] / lrow[r];
        }
}

// ---------------------------------------------------------------------------
extern "C" void kernel_launch(void* const* d_in, const int* in_sizes, int n_in,
                              void* d_out, int out_size, void* d_ws, size_t ws_size,
                              hipStream_t stream) {
    (void)in_sizes; (void)n_in; (void)out_size; (void)ws_size;
    const int*   positions = (const int*)d_in[0];
    const float* hidden    = (const float*)d_in[1];
    const float* w_qkv     = (const float*)d_in[2];
    const float* b_qkv     = (const float*)d_in[3];
    const float* w_o       = (const float*)d_in[4];
    const float* b_o       = (const float*)d_in[5];
    float* out = (float*)d_out;

    const size_t M = (size_t)NB * NS;                 // 4096
    // workspace regions (bytes), 256-aligned sizes
    char* ws = (char*)d_ws;
    size_t szA = M * NHID * 2;                        // hidden_bf16 / attn_bf16
    size_t szB = (size_t)OQKV * NHID * 2;             // wqkv_bf16 / wo_bf16
    size_t szC = M * OQKV * 4;                        // qkv_f32 / attn_f32
    size_t szD = M * QSZ * 2;                         // q_bf16
    size_t szE = M * KVSZ * 2;                        // k_bf16
    unsigned short* hbf  = (unsigned short*)(ws);
    unsigned short* wbf  = (unsigned short*)(ws + szA);
    float*          qkvf = (float*)(ws + szA + szB);
    unsigned short* qb   = (unsigned short*)(ws + szA + szB + szC);
    unsigned short* kb   = (unsigned short*)(ws + szA + szB + szC + szD);
    unsigned short* vb   = (unsigned short*)(ws + szA + szB + szC + szD + szE);
    float*          attnf = qkvf;                     // reuse after rope/split
    unsigned short* abf   = hbf;                      // reuse after qkv gemm
    unsigned short* wobf  = wbf;                      // reuse after qkv gemm

    auto cgrid = [](size_t n) {
        size_t g = (n + 255) / 256;
        return (unsigned)(g > 65535 ? 65535 : g);
    };

    // 1) convert hidden + w_qkv to bf16
    convert_f32_bf16<<<cgrid(M * NHID), 256, 0, stream>>>(hidden, hbf, M * NHID);
    convert_f32_bf16<<<cgrid((size_t)OQKV * NHID), 256, 0, stream>>>(
        w_qkv, wbf, (size_t)OQKV * NHID);
    // 2) qkv = hidden @ w_qkv^T + b_qkv    (M=4096, N=5632, K=4608)
    gemm_bf16_kernel<<<dim3(OQKV / GBN, (unsigned)(M / GBM)), 256, 0, stream>>>(
        hbf, wbf, b_qkv, qkvf, (int)M, OQKV, NHID);
    // 3) RoPE + split -> bf16 q/k/v
    {
        size_t n = M * (size_t)(NHEADS * 64 + NKVH * 64 + NKVH * 128);
        rope_split_kernel<<<cgrid(n), 256, 0, stream>>>(qkvf, positions, qb, kb, vb);
    }
    // 4) causal GQA flash attention -> attn f32 (reuses qkv_f32 region)
    attn_kernel<<<dim3(NS / 128, NHEADS, NB), 256, 0, stream>>>(qb, kb, vb, attnf);
    // 5) convert attn + w_o to bf16 (reuse regions A/B)
    convert_f32_bf16<<<cgrid(M * QSZ), 256, 0, stream>>>(attnf, abf, M * QSZ);
    convert_f32_bf16<<<cgrid((size_t)NHID * QSZ), 256, 0, stream>>>(
        w_o, wobf, (size_t)NHID * QSZ);
    // 6) out = attn @ w_o^T + b_o          (M=4096, N=4608, K=4608)
    gemm_bf16_kernel<<<dim3(NHID / GBN, (unsigned)(M / GBM)), 256, 0, stream>>>(
        abf, wobf, b_o, out, (int)M, NHID, QSZ);
}
